// TemporalCrossTransformer_multihead_28071906247070
// MI455X (gfx1250) — compile-verified, matrix operand order checked by reading, not run on hardware
//
#include <hip/hip_runtime.h>
#include <hip/hip_bf16.h>

typedef __bf16 bf16_t;
typedef __attribute__((ext_vector_type(16))) __bf16 v16bf;
typedef __attribute__((ext_vector_type(8)))  __bf16 v8bf;
typedef __attribute__((ext_vector_type(8)))  float  v8f;

#define SEQ    8
#define DIN    2048
#define DOUT   1152
#define WAY    5
#define SHOT   5
#define NSUP   25
#define NQRY   100
#define NSAMP  125
#define TT     56      // C(8,3)
#define AROWS  1000    // NSAMP*SEQ
#define KDIM   2048
#define NB     6912    // 2 mats * 3 ts * DOUT
#define HALFNB 3456
#define QROWS  5600    // NQRY*TT
#define NKEY   280     // SHOT*TT
#define NKEYP  288
#define CVT_LD 1440    // WAY*NKEYP

#if defined(__AMDGCN__) && __has_builtin(__builtin_amdgcn_tensor_load_to_lds) && \
    __has_builtin(__builtin_amdgcn_s_wait_tensorcnt)
#define HAVE_TDM 1
#else
#define HAVE_TDM 0
#endif

// ---------------- activation + positional encoding -> bf16 ----------------
__global__ __launch_bounds__(256) void act_kernel(const float* __restrict__ sup,
                                                  const float* __restrict__ qry,
                                                  bf16_t* __restrict__ actg,
                                                  bf16_t* __restrict__ actl) {
  long idx = (long)blockIdx.x * 256 + threadIdx.x;
  if (idx >= (long)NSAMP * SEQ * DIN) return;
  int c = (int)(idx % DIN);
  int s = (int)((idx / DIN) % SEQ);
  int n = (int)(idx / ((long)DIN * SEQ));
  float freq = expf(-9.210340371976184f * (float)(c & ~1) / (float)DIN);
  float ang  = (float)s * freq;
  float pe   = ((c & 1) ? cosf(ang) : sinf(ang)) * 0.1f;
  const float* src = (n < NSUP) ? (sup + ((long)n * SEQ + s) * (2 * DIN))
                                : (qry + ((long)(n - NSUP) * SEQ + s) * (2 * DIN));
  actg[idx] = (bf16_t)(src[c] + pe);
  actl[idx] = (bf16_t)(src[DIN + c] + pe);
}

// -------- fused-weight transpose: Bt[mat*3456+ts*1152+j][k] = W[ts*2048+k][j] --------
__global__ __launch_bounds__(256) void wtrans_kernel(const float* __restrict__ Wk,
                                                     const float* __restrict__ Wv,
                                                     const float* __restrict__ Wkl,
                                                     const float* __restrict__ Wvl,
                                                     bf16_t* __restrict__ Btg,
                                                     bf16_t* __restrict__ Btl) {
  int z = blockIdx.z;                 // stream*6 + mat*3 + ts
  int stream = z / 6, mat = (z % 6) / 3, ts = z % 3;
  const float* W = (stream == 0) ? (mat == 0 ? Wk : Wv) : (mat == 0 ? Wkl : Wvl);
  bf16_t* Bt = (stream == 0) ? Btg : Btl;
  __shared__ float tile[32][33];
  int j0 = blockIdx.x * 32;           // over DOUT (36 tiles)
  int k0 = blockIdx.y * 32;           // over KDIM (64 tiles)
  int tx = threadIdx.x & 31, ty = threadIdx.x >> 5;
#pragma unroll
  for (int yy = 0; yy < 4; ++yy) {
    int kk = k0 + ty + yy * 8;
    tile[ty + yy * 8][tx] = W[(long)(ts * KDIM + kk) * DOUT + j0 + tx];
  }
  __syncthreads();
#pragma unroll
  for (int yy = 0; yy < 4; ++yy) {
    int jj = ty + yy * 8;
    Bt[(long)(mat * HALFNB + ts * DOUT + j0 + jj) * KDIM + k0 + tx] = (bf16_t)tile[tx][jj];
  }
}

// ---------------- stable rank of support labels (grouping order) ----------------
__global__ void rank_kernel(const int* __restrict__ lab, int* __restrict__ rank) {
  int n = threadIdx.x;
  if (n < NSUP) {
    int r = 0;
    for (int i = 0; i < NSUP; ++i)
      if (lab[i] < lab[n] || (lab[i] == lab[n] && i < n)) r++;
    rank[n] = r;
  }
}

__global__ __launch_bounds__(256) void zero_bf16_kernel(bf16_t* __restrict__ p, long n) {
  long i = (long)blockIdx.x * 256 + threadIdx.x;
  if (i < n) p[i] = (bf16_t)0.0f;
}

// ---------------- generic NT bf16 WMMA GEMM: C = alpha * A(MxK) * B(NxK)^T ----------------
#define TM 128
#define TN 64
#define TKK 32
#define LDSA 40   // row stride (elements): 80B = 20 banks -> conflict-free frag reads

__device__ inline v16bf load_frag(const bf16_t* __restrict__ base, int koff) {
  // 16-bit operand layout (ISA 7.12.2): lane half selects k+8; element halves are k+0..7 / k+16..23
  v8bf a = *(const v8bf*)(base + koff);
  v8bf b = *(const v8bf*)(base + koff + 16);
  v16bf f;
#pragma unroll
  for (int i = 0; i < 8; ++i) { f[i] = a[i]; f[i + 8] = b[i]; }
  return f;
}

#if HAVE_TDM
typedef __attribute__((ext_vector_type(4))) unsigned int u32x4_t;
typedef __attribute__((ext_vector_type(8))) int i32x8_t;
typedef __attribute__((ext_vector_type(4))) int i32x4_t;

// 2D bf16 tile load via Tensor Data Mover: tile_rows x 32 elems, row stride ld (elems),
// rows beyond rows_rem zero-filled; LDS padded +4 DWORDs per 16 DWORDs -> 40-elem row stride.
__device__ inline void tdm_load_tile(unsigned lds_addr, const bf16_t* g,
                                     int ld, int rows_rem, int tile_rows) {
  unsigned long long ga = (unsigned long long)(size_t)g;
  u32x4_t g0;
  g0[0] = 1u;                                        // count=1, user mode
  g0[1] = lds_addr;                                  // LDS byte address
  g0[2] = (unsigned)ga;                              // global_addr[31:0]
  g0[3] = (unsigned)(ga >> 32) | (2u << 30);         // global_addr[56:32] | type=2
  i32x8_t g1;
  g1[0] = (int)((1u << 16)            // data_size = 2B
              | (1u << 20)            // pad_enable
              | (3u << 22)            // pad_interval: 16 DWORDs
              | (3u << 25));          // pad_amount:   4 DWORDs
  g1[1] = (int)(((unsigned)TKK & 0xffffu) << 16);                 // tensor_dim0 lo
  g1[2] = (int)(((unsigned)rows_rem & 0xffffu) << 16);            // dim0 hi=0 | tensor_dim1 lo
  g1[3] = (int)((((unsigned)rows_rem >> 16) & 0xffffu)
              | ((unsigned)TKK << 16));                           // dim1 hi | tile_dim0=32
  g1[4] = (int)((unsigned)tile_rows & 0xffffu);                   // tile_dim1 | tile_dim2=0
  g1[5] = (int)(unsigned)ld;                                      // tensor_dim0_stride lo
  g1[6] = 0;                                                      // stride hi | dim1_stride (2D: unused)
  g1[7] = 0;
  i32x4_t z4 = {};
#if __clang_major__ >= 23
  i32x8_t z8 = {};
  __builtin_amdgcn_tensor_load_to_lds(g0, g1, z4, z4, z8, 0);
#else
  __builtin_amdgcn_tensor_load_to_lds(g0, g1, z4, z4, 0);
#endif
}
#endif

__global__ __launch_bounds__(256) void gemm_bf16_nt(const bf16_t* __restrict__ A, int lda,
                                                    const bf16_t* __restrict__ B, int ldb,
                                                    float* __restrict__ C, int ldc,
                                                    int M, int N, int K, int Bclamp,
                                                    float alpha) {
  const int tid  = threadIdx.x;
  const int lane = tid & 31;
  const int wave = tid >> 5;
  const int mOff = (wave & 3) * 32;
  const int nOff = (wave >> 2) * 32;
  const int blockM = blockIdx.y * TM;
  const int blockN = blockIdx.x * TN;
  const int ksteps = K / TKK;

  v8f acc00 = {}, acc01 = {}, acc10 = {}, acc11 = {};

#if HAVE_TDM
  // -------- double-buffered TDM pipeline: DMA next tile while computing current --------
  __shared__ bf16_t As[2][TM * LDSA];
  __shared__ bf16_t Bs[2][TN * LDSA];
  const bf16_t* aTile = A + (long)blockM * lda;
  const bf16_t* bTile = B + (long)blockN * ldb;
  const int aRem = M - blockM;
  const int bRem = Bclamp - blockN;
  unsigned aAddr[2] = {(unsigned)(size_t)&As[0][0], (unsigned)(size_t)&As[1][0]};
  unsigned bAddr[2] = {(unsigned)(size_t)&Bs[0][0], (unsigned)(size_t)&Bs[1][0]};
  if (wave == 0) {
    tdm_load_tile(aAddr[0], aTile, lda, aRem, TM);
    tdm_load_tile(bAddr[0], bTile, ldb, bRem, TN);
  }
  for (int ks = 0; ks < ksteps; ++ks) {
    const int buf = ks & 1;
    if (wave == 0) {
      if (ks + 1 < ksteps) {
        tdm_load_tile(aAddr[buf ^ 1], aTile + (ks + 1) * TKK, lda, aRem, TM);
        tdm_load_tile(bAddr[buf ^ 1], bTile + (ks + 1) * TKK, ldb, bRem, TN);
        __builtin_amdgcn_s_wait_tensorcnt(2);   // retire current tile's 2 DMAs
      } else {
        __builtin_amdgcn_s_wait_tensorcnt(0);
      }
    }
    __syncthreads();                            // publish tile ks
    const int koff = (lane >= 16) ? 8 : 0;
    const int r = lane & 15;
    v16bf a0 = load_frag(&As[buf][(mOff + r) * LDSA],      koff);
    v16bf a1 = load_frag(&As[buf][(mOff + 16 + r) * LDSA], koff);
    v16bf b0 = load_frag(&Bs[buf][(nOff + r) * LDSA],      koff);
    v16bf b1 = load_frag(&Bs[buf][(nOff + 16 + r) * LDSA], koff);
    acc00 = __builtin_amdgcn_wmma_f32_16x16x32_bf16(false, a0, false, b0, (short)0, acc00, false, false);
    acc01 = __builtin_amdgcn_wmma_f32_16x16x32_bf16(false, a0, false, b1, (short)0, acc01, false, false);
    acc10 = __builtin_amdgcn_wmma_f32_16x16x32_bf16(false, a1, false, b0, (short)0, acc10, false, false);
    acc11 = __builtin_amdgcn_wmma_f32_16x16x32_bf16(false, a1, false, b1, (short)0, acc11, false, false);
    __syncthreads();                            // all readers done before buf is overwritten
  }
#else
  // -------- fallback: per-thread global loads + LDS staging --------
  __shared__ bf16_t As[TM * LDSA];
  __shared__ bf16_t Bs[TN * LDSA];
  const int arow = tid >> 1;
  const int acol = (tid & 1) * 16;
  const int agrow = (blockM + arow < M) ? (blockM + arow) : (M - 1);
  const int brow = tid >> 2;
  const int bcol = (tid & 3) * 8;
  const int bgrow = (blockN + brow < Bclamp) ? (blockN + brow) : (Bclamp - 1);
  const bf16_t* aSrc = A + (long)agrow * lda + acol;
  const bf16_t* bSrc = B + (long)bgrow * ldb + bcol;
  for (int ks = 0; ks < ksteps; ++ks) {
    const int k0 = ks * TKK;
    if (ks + 1 < ksteps) {
      __builtin_prefetch(aSrc + k0 + TKK, 0, 1);
      __builtin_prefetch(bSrc + k0 + TKK, 0, 1);
    }
    {
      v8bf a0 = *(const v8bf*)(aSrc + k0);
      v8bf a1 = *(const v8bf*)(aSrc + k0 + 8);
      *(v8bf*)(&As[arow * LDSA + acol])     = a0;
      *(v8bf*)(&As[arow * LDSA + acol + 8]) = a1;
    }
    {
      v8bf b0 = *(const v8bf*)(bSrc + k0);
      *(v8bf*)(&Bs[brow * LDSA + bcol]) = b0;
    }
    __syncthreads();
    const int koff = (lane >= 16) ? 8 : 0;
    const int r = lane & 15;
    v16bf a0 = load_frag(&As[(mOff + r) * LDSA],      koff);
    v16bf a1 = load_frag(&As[(mOff + 16 + r) * LDSA], koff);
    v16bf b0 = load_frag(&Bs[(nOff + r) * LDSA],      koff);
    v16bf b1 = load_frag(&Bs[(nOff + 16 + r) * LDSA], koff);
    acc00 = __builtin_amdgcn_wmma_f32_16x16x32_bf16(false, a0, false, b0, (short)0, acc00, false, false);
    acc01 = __builtin_amdgcn_wmma_f32_16x16x32_bf16(false, a0, false, b1, (short)0, acc01, false, false);
    acc10 = __builtin_amdgcn_wmma_f32_16x16x32_bf16(false, a1, false, b0, (short)0, acc10, false, false);
    acc11 = __builtin_amdgcn_wmma_f32_16x16x32_bf16(false, a1, false, b1, (short)0, acc11, false, false);
    __syncthreads();
  }
#endif

  // C layout (ISA 7.12.2): VGPR r -> M = r + (lane>=16 ? 8 : 0); N = lane&15
  const int radd = (lane >= 16) ? 8 : 0;
  const int ccol = blockN + nOff + (lane & 15);
#pragma unroll
  for (int mi = 0; mi < 2; ++mi) {
#pragma unroll
    for (int ni = 0; ni < 2; ++ni) {
      v8f acc = (mi == 0) ? (ni == 0 ? acc00 : acc01) : (ni == 0 ? acc10 : acc11);
      int colw = ccol + ni * 16;
      if (colw < N) {
#pragma unroll
        for (int r2 = 0; r2 < 8; ++r2) {
          int roww = blockM + mOff + mi * 16 + radd + r2;
          if (roww < M) C[(long)roww * ldc + colw] = acc[r2] * alpha;
        }
      }
    }
  }
}

// -------- combine: 3-way add of per-seq projections + bias, LN(K), scatter K/V --------
__global__ __launch_bounds__(256) void combine_kernel(const float* __restrict__ Y,
                                                      const float* __restrict__ bk,
                                                      const float* __restrict__ bv,
                                                      const float* __restrict__ lng,
                                                      const float* __restrict__ lnb,
                                                      const int* __restrict__ rank,
                                                      bf16_t* __restrict__ qk,
                                                      bf16_t* __restrict__ ck,
                                                      bf16_t* __restrict__ cvT,
                                                      float* __restrict__ qv) {
  __shared__ float krow[DOUT];
  __shared__ float s1[256], s2[256];
  int b = blockIdx.x;
  int n = b / TT;
  int t = b % TT;
  int i0 = 0, i1 = 1, i2 = 2;
  {
    int idx = 0;
    for (int a = 0; a < SEQ - 2; ++a)
      for (int bb = a + 1; bb < SEQ - 1; ++bb)
        for (int c = bb + 1; c < SEQ; ++c) {
          if (idx == t) { i0 = a; i1 = bb; i2 = c; }
          idx++;
        }
  }
  long y0 = (long)(n * SEQ + i0) * NB;
  long y1 = (long)(n * SEQ + i1) * NB;
  long y2 = (long)(n * SEQ + i2) * NB;
  bool isQ = (n >= NSUP);
  int gr = 0, w = 0, sh = 0, qn = 0;
  if (isQ) qn = n - NSUP;
  else { gr = rank[n]; w = gr / SHOT; sh = gr % SHOT; }

  float lsum = 0.f, lsq = 0.f;
  for (int d = threadIdx.x; d < DOUT; d += 256) {
    float kv = Y[y0 + d] + Y[y1 + DOUT + d] + Y[y2 + 2 * DOUT + d] + bk[d];
    float vv = Y[y0 + HALFNB + d] + Y[y1 + HALFNB + DOUT + d] + Y[y2 + HALFNB + 2 * DOUT + d] + bv[d];
    krow[d] = kv;
    lsum += kv; lsq += kv * kv;
    if (isQ) qv[((long)qn * TT + t) * DOUT + d] = vv;
    else     cvT[(long)d * CVT_LD + w * NKEYP + sh * TT + t] = (bf16_t)vv;
  }
  s1[threadIdx.x] = lsum; s2[threadIdx.x] = lsq;
  __syncthreads();
  for (int s = 128; s > 0; s >>= 1) {
    if (threadIdx.x < s) { s1[threadIdx.x] += s1[threadIdx.x + s]; s2[threadIdx.x] += s2[threadIdx.x + s]; }
    __syncthreads();
  }
  float mean = s1[0] / (float)DOUT;
  float var  = s2[0] / (float)DOUT - mean * mean;
  float inv  = rsqrtf(var + 1e-5f);
  for (int d = threadIdx.x; d < DOUT; d += 256) {
    float ln = (krow[d] - mean) * inv * lng[d] + lnb[d];
    if (isQ) qk[((long)qn * TT + t) * DOUT + d] = (bf16_t)ln;
    else     ck[((long)(gr * TT + t)) * DOUT + d] = (bf16_t)ln;
  }
}

// ---------------- softmax over 280 keys, pad to 288, emit bf16 ----------------
__global__ __launch_bounds__(128) void softmax_kernel(const float* __restrict__ L,
                                                      bf16_t* __restrict__ P) {
  int row = blockIdx.x;
  const float* lr = L + (long)row * NKEYP;
  __shared__ float sh[128];
  float m = -1e30f;
  for (int c = threadIdx.x; c < NKEY; c += 128) m = fmaxf(m, lr[c]);
  sh[threadIdx.x] = m; __syncthreads();
  for (int s = 64; s > 0; s >>= 1) {
    if (threadIdx.x < s) sh[threadIdx.x] = fmaxf(sh[threadIdx.x], sh[threadIdx.x + s]);
    __syncthreads();
  }
  m = sh[0]; __syncthreads();
  float sum = 0.f;
  for (int c = threadIdx.x; c < NKEY; c += 128) sum += expf(lr[c] - m);
  sh[threadIdx.x] = sum; __syncthreads();
  for (int s = 64; s > 0; s >>= 1) {
    if (threadIdx.x < s) sh[threadIdx.x] += sh[threadIdx.x + s];
    __syncthreads();
  }
  float inv = 1.f / sh[0];
  bf16_t* pr = P + (long)row * NKEYP;
  for (int c = threadIdx.x; c < NKEYP; c += 128)
    pr[c] = (c < NKEY) ? (bf16_t)(expf(lr[c] - m) * inv) : (bf16_t)0.0f;
}

// ---------------- fused squared-distance per (way, query) ----------------
__global__ __launch_bounds__(256) void dist_kernel(const float* __restrict__ qvg,
                                                   const float* __restrict__ qvl,
                                                   const float* __restrict__ pg,
                                                   const float* __restrict__ pl,
                                                   float* __restrict__ out, int w) {
  int q = blockIdx.x;
  long base = (long)q * TT * DOUT;
  float acc = 0.f;
  for (int i = threadIdx.x; i < TT * DOUT; i += 256) {
    float dg = qvg[base + i] - pg[base + i];
    float dl = qvl[base + i] - pl[base + i];
    acc += dg * dg + dl * dl;
  }
  __shared__ float sh[256];
  sh[threadIdx.x] = acc; __syncthreads();
  for (int s = 128; s > 0; s >>= 1) {
    if (threadIdx.x < s) sh[threadIdx.x] += sh[threadIdx.x + s];
    __syncthreads();
  }
  if (threadIdx.x == 0) out[q * WAY + w] = -sh[0] / (float)TT;
}

extern "C" void kernel_launch(void* const* d_in, const int* in_sizes, int n_in,
                              void* d_out, int out_size, void* d_ws, size_t ws_size,
                              hipStream_t stream) {
  (void)in_sizes; (void)n_in; (void)out_size; (void)ws_size;
  const float* sup  = (const float*)d_in[0];
  const float* qry  = (const float*)d_in[1];
  const int*   lab  = (const int*)d_in[2];
  const float* Wk   = (const float*)d_in[3];
  const float* bk   = (const float*)d_in[4];
  const float* Wv   = (const float*)d_in[5];
  const float* bv   = (const float*)d_in[6];
  const float* Wkl  = (const float*)d_in[7];
  const float* bkl  = (const float*)d_in[8];
  const float* Wvl  = (const float*)d_in[9];
  const float* bvl  = (const float*)d_in[10];
  const float* lng  = (const float*)d_in[11];
  const float* lnb  = (const float*)d_in[12];
  float* out = (float*)d_out;

  char* p = (char*)d_ws;
  auto alloc = [&](size_t bytes) -> void* {
    void* r = (void*)p;
    p += (bytes + 255) & ~(size_t)255;
    return r;
  };
  bf16_t* actg = (bf16_t*)alloc((size_t)AROWS * KDIM * 2);
  bf16_t* actl = (bf16_t*)alloc((size_t)AROWS * KDIM * 2);
  bf16_t* Btg  = (bf16_t*)alloc((size_t)NB * KDIM * 2);
  bf16_t* Btl  = (bf16_t*)alloc((size_t)NB * KDIM * 2);
  float*  Yg   = (float*) alloc((size_t)AROWS * NB * 4);
  float*  Yl   = (float*) alloc((size_t)AROWS * NB * 4);
  bf16_t* qkg  = (bf16_t*)alloc((size_t)QROWS * DOUT * 2);
  bf16_t* qkl  = (bf16_t*)alloc((size_t)QROWS * DOUT * 2);
  bf16_t* ckg  = (bf16_t*)alloc((size_t)NSUP * TT * DOUT * 2);
  bf16_t* ckl  = (bf16_t*)alloc((size_t)NSUP * TT * DOUT * 2);
  bf16_t* cvTg = (bf16_t*)alloc((size_t)DOUT * CVT_LD * 2);
  bf16_t* cvTl = (bf16_t*)alloc((size_t)DOUT * CVT_LD * 2);
  float*  qvg  = (float*) alloc((size_t)QROWS * DOUT * 4);
  float*  qvl  = (float*) alloc((size_t)QROWS * DOUT * 4);
  float*  lgt  = (float*) alloc((size_t)QROWS * NKEYP * 4);
  bf16_t* Pbf  = (bf16_t*)alloc((size_t)QROWS * NKEYP * 2);
  float*  prg  = (float*) alloc((size_t)QROWS * DOUT * 4);
  float*  prl  = (float*) alloc((size_t)QROWS * DOUT * 4);
  int*    rank = (int*)   alloc(NSUP * 4);

  const float scale = 0.029462782549439483f; // 1/sqrt(1152)

  // 1) activations + PE -> bf16
  {
    long n = (long)NSAMP * SEQ * DIN;
    act_kernel<<<dim3((unsigned)((n + 255) / 256)), dim3(256), 0, stream>>>(sup, qry, actg, actl);
  }
  // 2) fused weight transpose -> bf16 (NT layout)
  wtrans_kernel<<<dim3(36, 64, 12), dim3(256), 0, stream>>>(Wk, Wv, Wkl, Wvl, Btg, Btl);
  // 3) support grouping ranks
  rank_kernel<<<dim3(1), dim3(32), 0, stream>>>(lab, rank);
  // 4) zero cvT padding slots
  {
    long n = (long)DOUT * CVT_LD;
    zero_bf16_kernel<<<dim3((unsigned)((n + 255) / 256)), dim3(256), 0, stream>>>(cvTg, n);
    zero_bf16_kernel<<<dim3((unsigned)((n + 255) / 256)), dim3(256), 0, stream>>>(cvTl, n);
  }
  // 5) projection GEMMs: Y(1000 x 6912) = act(1000 x 2048) * Bt(6912 x 2048)^T
  {
    dim3 grid((NB + TN - 1) / TN, (AROWS + TM - 1) / TM);
    gemm_bf16_nt<<<grid, dim3(256), 0, stream>>>(actg, KDIM, Btg, KDIM, Yg, NB, AROWS, NB, KDIM, NB, 1.0f);
    gemm_bf16_nt<<<grid, dim3(256), 0, stream>>>(actl, KDIM, Btl, KDIM, Yl, NB, AROWS, NB, KDIM, NB, 1.0f);
  }
  // 6) combine (tuple sum + bias + LN) -> qk/ck bf16, qv f32, cv^T bf16
  combine_kernel<<<dim3(NSAMP * TT), dim3(256), 0, stream>>>(Yg, bk, bv, lng, lnb, rank, qkg, ckg, cvTg, qvg);
  combine_kernel<<<dim3(NSAMP * TT), dim3(256), 0, stream>>>(Yl, bkl, bvl, lng, lnb, rank, qkl, ckl, cvTl, qvl);

  // 7) attention per (way, stream) + fused distance per way
  for (int w = 0; w < WAY; ++w) {
    for (int s = 0; s < 2; ++s) {
      const bf16_t* qk  = s ? qkl : qkg;
      const bf16_t* ck  = s ? ckl : ckg;
      const bf16_t* cvT = s ? cvTl : cvTg;
      float* proto      = s ? prl : prg;
      // logits(5600 x 288) = qk(5600 x 1152) * ck_way(280 x 1152)^T * scale
      {
        dim3 grid((NKEYP + TN - 1) / TN, (QROWS + TM - 1) / TM);
        gemm_bf16_nt<<<grid, dim3(256), 0, stream>>>(qk, DOUT, ck + (long)w * NKEY * DOUT, DOUT,
                                                     lgt, NKEYP, QROWS, NKEYP, DOUT, NKEY, scale);
      }
      softmax_kernel<<<dim3(QROWS), dim3(128), 0, stream>>>(lgt, Pbf);
      // proto(5600 x 1152) = P(5600 x 288) * cvT_way(1152 x 288)^T
      {
        dim3 grid((DOUT + TN - 1) / TN, (QROWS + TM - 1) / TM);
        gemm_bf16_nt<<<grid, dim3(256), 0, stream>>>(Pbf, NKEYP, cvT + (long)w * NKEYP, CVT_LD,
                                                     proto, DOUT, QROWS, DOUT, NKEYP, DOUT, 1.0f);
      }
    }
    dist_kernel<<<dim3(NQRY), dim3(256), 0, stream>>>(qvg, qvl, prg, prl, out, w);
  }
}
